// Net_17257178595369
// MI455X (gfx1250) — matrix operand order repaired and checked by reading, not compiled, not run on hardware
//
#include <hip/hip_runtime.h>
#include <hip/hip_bf16.h>
#include <stdint.h>

#define L 32
#define DFdim 128

typedef _Float16 h16 __attribute__((ext_vector_type(16)));
typedef _Float16 h8  __attribute__((ext_vector_type(8)));
typedef float    f8  __attribute__((ext_vector_type(8)));

// ---------------- f32 -> f16 convert ----------------
__global__ void k_cvt_f16(const float* __restrict__ src, _Float16* __restrict__ dst, int n) {
  int i = blockIdx.x * blockDim.x + threadIdx.x;
  if (i < n) dst[i] = (_Float16)src[i];
}

// src [K x 32] row-major f32  ->  dst [32 x K] f16 (transposed, for vectorized B loads)
__global__ void k_cvt_f16_T(const float* __restrict__ src, _Float16* __restrict__ dst, int K) {
  int i = blockIdx.x * blockDim.x + threadIdx.x;
  if (i < K * 32) {
    int k = i >> 5, n = i & 31;
    dst[n * K + k] = (_Float16)src[i];
  }
}

// ---------------- WMMA GEMM: C[M x 32] = A[M x K] @ B[K x 32], K % 32 == 0 --------------
// Bt is B transposed: Bt[n*K + k].  One wave computes a 16-row tile, both 16-col n-tiles.
__global__ void k_wmma_gemm(const _Float16* __restrict__ A, const _Float16* __restrict__ Bt,
                            float* __restrict__ C, int M, int K) {
  const int lane = threadIdx.x & 31;
  const int tile = blockIdx.x * (blockDim.x >> 5) + (threadIdx.x >> 5);
  const int m0 = tile * 16;
  if (m0 >= M) return;                       // wave-uniform
  const int hf = lane >> 4;
  const int n  = lane & 15;
  const int arow = (m0 + n < M) ? (m0 + n) : (M - 1);
  const _Float16* a_base  = A  + (size_t)arow * K + hf * 16;
  const _Float16* b0_base = Bt + (size_t)n        * K + hf * 16;
  const _Float16* b1_base = Bt + (size_t)(n + 16) * K + hf * 16;
  f8 c0 = {}; f8 c1 = {};
  for (int k0 = 0; k0 < K; k0 += 32) {
    // A fragment (16-bit 16x32): lane M=n; elem i -> K=k0+hf*16+i (matches B).
    // Two 16B vector loads.
    h8 alo = *(const h8*)(a_base + k0);
    h8 ahi = *(const h8*)(a_base + k0 + 8);
    h16 a;
#pragma unroll
    for (int i = 0; i < 8; ++i) { a[i] = alo[i]; a[8 + i] = ahi[i]; }
    // B fragment (16-bit 32x16): lane N=n(+16); elem i -> K=k0+hf*16+i.  Contiguous 32B.
    h16 b0 = *(const h16*)(b0_base + k0);
    h16 b1 = *(const h16*)(b1_base + k0);
    c0 = __builtin_amdgcn_wmma_f32_16x16x32_f16(false, a, false, b0, (short)0, c0, false, false);
    c1 = __builtin_amdgcn_wmma_f32_16x16x32_f16(false, a, false, b1, (short)0, c1, false, false);
  }
  // C/D 16x16 f32: VGPR v -> M = hf*8 + v, N = lane&15
  if (m0 + 16 <= M) {                         // full tile: unguarded stores (wave-uniform test)
#pragma unroll
    for (int v = 0; v < 8; ++v) {
      const int m = m0 + hf * 8 + v;
      C[(size_t)m * 32 + n]      = c0[v];
      C[(size_t)m * 32 + n + 16] = c1[v];
    }
  } else {
#pragma unroll
    for (int v = 0; v < 8; ++v) {
      const int m = m0 + hf * 8 + v;
      if (m < M) {
        C[(size_t)m * 32 + n]      = c0[v];
        C[(size_t)m * 32 + n + 16] = c1[v];
      }
    }
  }
}

// ---------------- GCN pieces ----------------
__global__ void k_deg(const int* __restrict__ col, float* __restrict__ deg, int E) {
  int e = blockIdx.x * blockDim.x + threadIdx.x;
  if (e < E) atomicAdd(&deg[col[e]], 1.0f);
}
__global__ void k_dinv(const float* __restrict__ deg, float* __restrict__ dinv, int N) {
  int i = blockIdx.x * blockDim.x + threadIdx.x;
  if (i < N) dinv[i] = rsqrtf(deg[i] + 1.0f);   // +1 self loop; always > 0
}
// out[n,c] = dinv[n]^2 * h[n,c] + bias[c]   (self-loop term + bias)
__global__ void k_agg_init(const float* __restrict__ h, const float* __restrict__ dinv,
                           const float* __restrict__ bias, float* __restrict__ out, int N) {
  int i = blockIdx.x * blockDim.x + threadIdx.x;
  if (i < N * L) {
    int nn = i >> 5, c = i & 31;
    float d = dinv[nn];
    out[i] = d * d * h[i] + bias[c];
  }
}
// out[col[e],c] += dinv[row]*dinv[col]*h[row[e],c]
__global__ void k_agg_edge(const float* __restrict__ h, const float* __restrict__ dinv,
                           const int* __restrict__ row, const int* __restrict__ col,
                           float* __restrict__ out, int E) {
  int i = blockIdx.x * blockDim.x + threadIdx.x;
  if (i < E * L) {
    int e = i >> 5, c = i & 31;
    int r = row[e], cc = col[e];
    atomicAdd(&out[(size_t)cc * L + c], dinv[r] * dinv[cc] * h[(size_t)r * L + c]);
  }
}

// ---------------- needed-edge inverse maps ----------------
__global__ void k_slots(const int* __restrict__ pred, const int* __restrict__ perm,
                        int* __restrict__ needSlot, int* __restrict__ slotEdge, int P, int EM) {
  int q = blockIdx.x * blockDim.x + threadIdx.x;
  if (q >= P) return;
  int e = pred[q];
  if ((unsigned)e < (unsigned)EM) {
    needSlot[e] = q;            // duplicate writers: arbitrary winner is consistent for all readers
    slotEdge[q] = e;
    int e2 = perm[e];
    if ((unsigned)e2 < (unsigned)EM) { needSlot[e2] = P + q; slotEdge[P + q] = e2; }
    else slotEdge[P + q] = -1;
  } else {
    slotEdge[q] = -1; slotEdge[P + q] = -1;
  }
}
__global__ void k_eflag(const int* __restrict__ idxE, unsigned char* __restrict__ eflag, int E) {
  int i = blockIdx.x * blockDim.x + threadIdx.x;
  if (i < E) eflag[idxE[i]] = 1;
}

// ---------------- triangle products, accumulated only for needed merged edges ----------
// prod[t,c] = (A1[row[e1],c]+A2[col[e1],c]+bm1[c]) * (B1[row[e2],c]+B2[col[e2],c]+bm2[c])
__global__ void k_triples(const int* __restrict__ t1, const int* __restrict__ t2,
                          const int* __restrict__ td, const int* __restrict__ idxC,
                          const int* __restrict__ row, const int* __restrict__ col,
                          const float* __restrict__ A1, const float* __restrict__ A2,
                          const float* __restrict__ B1, const float* __restrict__ B2,
                          const float* __restrict__ bm1, const float* __restrict__ bm2,
                          const int* __restrict__ needSlot, float* __restrict__ acc, int T) {
  int t = blockIdx.x * blockDim.x + threadIdx.x;
  if (t >= T) return;
  int s = needSlot[idxC[td[t]]];
  if (s < 0) return;                               // vast majority early-exit
  int e1 = t1[t], e2 = t2[t];
  const float* a1 = A1 + (size_t)row[e1] * L;
  const float* a2 = A2 + (size_t)col[e1] * L;
  const float* b1 = B1 + (size_t)row[e2] * L;
  const float* b2 = B2 + (size_t)col[e2] * L;
  float* ap = acc + (size_t)s * L;
#pragma unroll 4
  for (int c = 0; c < L; ++c) {
    float xv = a1[c] + a2[c] + bm1[c];
    float mu = b1[c] + b2[c] + bm2[c];
    atomicAdd(&ap[c], xv * mu);
  }
}

// xm[s,c] = gemm_result + bm3[c] + (edge in original set ? Wm3[32,c] : 0)   (in place)
__global__ void k_xm(float* __restrict__ xm, const int* __restrict__ slotEdge,
                     const unsigned char* __restrict__ eflag,
                     const float* __restrict__ bm3, const float* __restrict__ Wm3, int S) {
  int i = blockIdx.x * blockDim.x + threadIdx.x;
  if (i >= S * L) return;
  int s = i >> 5, c = i & 31;
  int e = slotEdge[s];
  float v = 0.f;
  if (e >= 0) {
    v = xm[i] + bm3[c];
    if (eflag[e]) v += Wm3[32 * 32 + c];           // last row of Wm3 (indicator column)
  }
  xm[i] = v;
}

// out[q] = sum_c xm[s1,c]*xm[s2,c]*Wd[c] + sum_c x2[i0,c]*x2[i1,c]*Wd[32+c] + bd
__global__ void k_out(const float* __restrict__ xm, const float* __restrict__ x2,
                      const int* __restrict__ pred, const int* __restrict__ perm,
                      const int* __restrict__ needSlot, const int* __restrict__ pos,
                      const float* __restrict__ Wd, const float* __restrict__ bd,
                      float* __restrict__ out, int P, int EM) {
  int q = blockIdx.x * blockDim.x + threadIdx.x;
  if (q >= P) return;
  float r = bd[0];
  int e = pred[q];
  if ((unsigned)e < (unsigned)EM) {
    int s1 = needSlot[e];
    int e2 = perm[e];
    int s2 = ((unsigned)e2 < (unsigned)EM) ? needSlot[e2] : -1;
    if (s1 >= 0 && s2 >= 0) {
      const float* xa = xm + (size_t)s1 * L;
      const float* xb = xm + (size_t)s2 * L;
#pragma unroll 4
      for (int c = 0; c < L; ++c) r += xa[c] * xb[c] * Wd[c];
    }
  }
  const float* p0 = x2 + (size_t)pos[2 * q] * L;
  const float* p1 = x2 + (size_t)pos[2 * q + 1] * L;
#pragma unroll 4
  for (int c = 0; c < L; ++c) r += p0[c] * p1[c] * Wd[32 + c];
  out[q] = r;
}

// ---------------- host ----------------
static inline unsigned cdiv(long long a, long long b) { return (unsigned)((a + b - 1) / b); }

extern "C" void kernel_launch(void* const* d_in, const int* in_sizes, int n_in,
                              void* d_out, int out_size, void* d_ws, size_t ws_size,
                              hipStream_t stream) {
  const float* feat = (const float*)d_in[0];
  const float* W1  = (const float*)d_in[1];
  const float* b1  = (const float*)d_in[2];
  const float* W2  = (const float*)d_in[3];
  const float* b2  = (const float*)d_in[4];
  const float* Wm1 = (const float*)d_in[5];
  const float* bm1 = (const float*)d_in[6];
  const float* Wm2 = (const float*)d_in[7];
  const float* bm2 = (const float*)d_in[8];
  const float* Wm3 = (const float*)d_in[9];
  const float* bm3 = (const float*)d_in[10];
  const float* Wd  = (const float*)d_in[11];
  const float* bd  = (const float*)d_in[12];
  const int* row  = (const int*)d_in[13];
  const int* col  = (const int*)d_in[14];
  const int* pos  = (const int*)d_in[15];
  const int* t1   = (const int*)d_in[16];
  const int* t2   = (const int*)d_in[17];
  const int* td   = (const int*)d_in[18];
  const int* idxC = (const int*)d_in[19];
  const int* idxE = (const int*)d_in[20];
  const int* perm = (const int*)d_in[21];
  const int* pred = (const int*)d_in[22];

  const int N  = in_sizes[0] / DFdim;
  const int E  = in_sizes[13];
  const int T  = in_sizes[16];
  const int E2 = in_sizes[20];
  const int EM = in_sizes[21];
  const int P  = in_sizes[22];
  const int S  = 2 * P;

  // ---- workspace carve ----
  char* w = (char*)d_ws;
  auto carve = [&](size_t bytes) { char* p = w; w += (bytes + 255) & ~(size_t)255; return p; };
  float* deg   = (float*)carve((size_t)N * 4);
  float* dinv  = (float*)carve((size_t)N * 4);
  float* hbuf  = (float*)carve((size_t)N * L * 4);
  float* x1    = (float*)carve((size_t)N * L * 4);
  float* x2    = (float*)carve((size_t)N * L * 4);
  float* A1    = (float*)carve((size_t)N * L * 4);
  float* A2    = (float*)carve((size_t)N * L * 4);
  float* B1    = (float*)carve((size_t)N * L * 4);
  float* B2    = (float*)carve((size_t)N * L * 4);
  _Float16* feat_h = (_Float16*)carve((size_t)N * DFdim * 2);
  _Float16* x1h  = (_Float16*)carve((size_t)N * L * 2);
  _Float16* x2h  = (_Float16*)carve((size_t)N * L * 2);
  _Float16* W1hT  = (_Float16*)carve((size_t)DFdim * L * 2);   // [32 x 128]
  _Float16* W2hT  = (_Float16*)carve((size_t)L * L * 2);       // [32 x 32]
  _Float16* Wm1hT = (_Float16*)carve((size_t)2 * L * L * 2);   // top block, then bottom block
  _Float16* Wm2hT = (_Float16*)carve((size_t)2 * L * L * 2);
  _Float16* Wm3hT = (_Float16*)carve((size_t)L * L * 2);       // top 32 rows only
  int* needSlot  = (int*)carve((size_t)EM * 4);
  unsigned char* eflag = (unsigned char*)carve((size_t)EM);
  int* slotEdge  = (int*)carve((size_t)S * 4);
  float* acc     = (float*)carve((size_t)S * L * 4);
  _Float16* acch = (_Float16*)carve((size_t)S * L * 2);
  float* xmb     = (float*)carve((size_t)S * L * 4);

  // ---- init scratch (graph-capture-safe async memsets) ----
  hipMemsetAsync(deg, 0, (size_t)N * 4, stream);
  hipMemsetAsync(acc, 0, (size_t)S * L * 4, stream);
  hipMemsetAsync(needSlot, 0xFF, (size_t)EM * 4, stream);     // -1
  hipMemsetAsync(eflag, 0, (size_t)EM, stream);

  const int TB = 256;
  // degrees / norm
  k_deg<<<cdiv(E, TB), TB, 0, stream>>>(col, deg, E);
  k_dinv<<<cdiv(N, TB), TB, 0, stream>>>(deg, dinv, N);

  // f16 copies: activations plain, weights transposed (for contiguous B-fragment loads)
  k_cvt_f16<<<cdiv((long long)N * DFdim, TB), TB, 0, stream>>>(feat, feat_h, N * DFdim);
  k_cvt_f16_T<<<cdiv(DFdim * L, TB), TB, 0, stream>>>(W1, W1hT, DFdim);
  k_cvt_f16_T<<<cdiv(L * L, TB), TB, 0, stream>>>(W2, W2hT, L);
  k_cvt_f16_T<<<cdiv(L * L, TB), TB, 0, stream>>>(Wm1,          Wm1hT,          L); // top rows
  k_cvt_f16_T<<<cdiv(L * L, TB), TB, 0, stream>>>(Wm1 + L * L,  Wm1hT + L * L,  L); // bottom rows
  k_cvt_f16_T<<<cdiv(L * L, TB), TB, 0, stream>>>(Wm2,          Wm2hT,          L);
  k_cvt_f16_T<<<cdiv(L * L, TB), TB, 0, stream>>>(Wm2 + L * L,  Wm2hT + L * L,  L);
  k_cvt_f16_T<<<cdiv(L * L, TB), TB, 0, stream>>>(Wm3, Wm3hT, L);

  const int WPB = TB / 32;                    // 8 waves (= 8 row-tiles) per block
  unsigned gN = cdiv(cdiv(N, 16), WPB);
  unsigned gS = cdiv(cdiv(S, 16), WPB);

  // GCN layer 1
  k_wmma_gemm<<<gN, TB, 0, stream>>>(feat_h, W1hT, hbuf, N, DFdim);
  k_agg_init<<<cdiv((long long)N * L, TB), TB, 0, stream>>>(hbuf, dinv, b1, x1, N);
  k_agg_edge<<<cdiv((long long)E * L, TB), TB, 0, stream>>>(hbuf, dinv, row, col, x1, E);
  // GCN layer 2
  k_cvt_f16<<<cdiv((long long)N * L, TB), TB, 0, stream>>>(x1, x1h, N * L);
  k_wmma_gemm<<<gN, TB, 0, stream>>>(x1h, W2hT, hbuf, N, L);
  k_agg_init<<<cdiv((long long)N * L, TB), TB, 0, stream>>>(hbuf, dinv, b2, x2, N);
  k_agg_edge<<<cdiv((long long)E * L, TB), TB, 0, stream>>>(hbuf, dinv, row, col, x2, E);

  // per-node projections replacing the E x 64 x 32 gathered GEMMs
  k_cvt_f16<<<cdiv((long long)N * L, TB), TB, 0, stream>>>(x2, x2h, N * L);
  k_wmma_gemm<<<gN, TB, 0, stream>>>(x2h, Wm1hT,          A1, N, L);
  k_wmma_gemm<<<gN, TB, 0, stream>>>(x2h, Wm1hT + L * L,  A2, N, L);
  k_wmma_gemm<<<gN, TB, 0, stream>>>(x2h, Wm2hT,          B1, N, L);
  k_wmma_gemm<<<gN, TB, 0, stream>>>(x2h, Wm2hT + L * L,  B2, N, L);

  // inverse maps for needed merged edges, then sparse triangle accumulation
  k_slots<<<cdiv(P, TB), TB, 0, stream>>>(pred, perm, needSlot, slotEdge, P, EM);
  k_eflag<<<cdiv(E2, TB), TB, 0, stream>>>(idxE, eflag, E2);
  k_triples<<<cdiv(T, TB), TB, 0, stream>>>(t1, t2, td, idxC, row, col,
                                            A1, A2, B1, B2, bm1, bm2, needSlot, acc, T);

  // Wm3 projection of accumulated rows (dense WMMA GEMM over 2P needed rows)
  k_cvt_f16<<<cdiv((long long)S * L, TB), TB, 0, stream>>>(acc, acch, S * L);
  k_wmma_gemm<<<gS, TB, 0, stream>>>(acch, Wm3hT, xmb, S, L);
  k_xm<<<cdiv((long long)S * L, TB), TB, 0, stream>>>(xmb, slotEdge, eflag, bm3, Wm3, S);

  // final gather + dot
  k_out<<<cdiv(P, TB), TB, 0, stream>>>(xmb, x2, pred, perm, needSlot, pos,
                                        Wd, bd, (float*)d_out, P, EM);
}
